// InvarianceLoss_9208409883401
// MI455X (gfx1250) — matrix operand order repaired and checked by reading, not compiled, not run on hardware
//
#include <hip/hip_runtime.h>
#include <math.h>

// InvarianceLoss for MI455X (gfx1250, wave32).
// Pipeline:
//   k_inter    : per-batch 16x16 argmax co-occurrence counts via v_wmma_f32_16x16x32_f16
//                (one-hot GEMM; exact integer counts in f32 accumulators)
//   k_hungarian: exact O(K^3) Jonker-Volgenant (f64) per (batch, direction)
//   k_loss     : per-(b,n) L2 distances with permuted gather + block tree reduction
//   k_final    : deterministic f64 reduction of block partials -> scalar loss

typedef __attribute__((ext_vector_type(16))) _Float16 v16h;
typedef __attribute__((ext_vector_type(8)))  float    v8f;

#define B_BATCH 8
#define K_OBJ 16

__device__ __forceinline__ int argmax16(const float4* __restrict__ q) {
    float best = -1e30f; int arg = 0;
#pragma unroll
    for (int i = 0; i < 4; ++i) {
        float4 r = q[i];
        if (r.x > best) { best = r.x; arg = 4*i + 0; }
        if (r.y > best) { best = r.y; arg = 4*i + 1; }
        if (r.z > best) { best = r.z; arg = 4*i + 2; }
        if (r.w > best) { best = r.w; arg = 4*i + 3; }
    }
    return arg;
}

// ---------------------------------------------------------------------------
// Kernel 1: inter[b][g][p] = #{ n : argmax(m1[b,n,:])==g && argmax(m2[b,n,:])==p }
// Implemented as one-hot GEMM with V_WMMA_F32_16X16X32_F16.
// A (16x32, MxK): lane half h=L/16, row m=L%16; VGPR v<4: K=8h+2v(+1); v>=4: K=16+8h+2(v-4)(+1)
// B (32x16, KxN): lane half h, col p=L%16; VGPR v<4: K=8h+2v(+1); v>=4: K=16+8h+2(v-4)(+1)
// C (16x16): lane half h, col p=L%16; VGPR r holds row m=r+8h
// ---------------------------------------------------------------------------
__global__ void k_inter(const float* __restrict__ m1, const float* __restrict__ m2,
                        float* __restrict__ inter, int N) {
    const int b    = blockIdx.y;
    const int lane = threadIdx.x & 31;
    const int waveInBlock   = threadIdx.x >> 5;
    const int wavesPerBlock = blockDim.x >> 5;
    const int waveId = blockIdx.x * wavesPerBlock + waveInBlock;
    const int nWaves = gridDim.x * wavesPerBlock;

    const size_t base = (size_t)b * (size_t)N * 16;
    const float* p1 = m1 + base;
    const float* p2 = m2 + base;
    const int h    = lane >> 4;
    const int mrow = lane & 15;

    v8f acc = {0.f, 0.f, 0.f, 0.f, 0.f, 0.f, 0.f, 0.f};
    const int nChunks = N >> 5;

    for (int c = waveId; c < nChunks; c += nWaves) {
        const int n = (c << 5) + lane;
        const float4* q1 = (const float4*)(p1 + (size_t)n * 16);
        const float4* q2 = (const float4*)(p2 + (size_t)n * 16);
        const int a1 = argmax16(q1);
        const int a2 = argmax16(q2);

        v16h A, Bm;
#pragma unroll
        for (int v = 0; v < 8; ++v) {
            const int ka = (v < 4) ? (8*h + 2*v) : (16 + 8*h + 2*(v - 4));
            const int a1k0 = __shfl(a1, ka,     32);
            const int a1k1 = __shfl(a1, ka + 1, 32);
            A[2*v]     = (a1k0 == mrow) ? (_Float16)1.0f : (_Float16)0.0f;
            A[2*v + 1] = (a1k1 == mrow) ? (_Float16)1.0f : (_Float16)0.0f;

            const int kb = (v < 4) ? (8*h + 2*v) : (16 + 8*h + 2*(v - 4));
            const int a2k0 = __shfl(a2, kb,     32);
            const int a2k1 = __shfl(a2, kb + 1, 32);
            Bm[2*v]     = (a2k0 == mrow) ? (_Float16)1.0f : (_Float16)0.0f;
            Bm[2*v + 1] = (a2k1 == mrow) ? (_Float16)1.0f : (_Float16)0.0f;
        }
        acc = __builtin_amdgcn_wmma_f32_16x16x32_f16(
            /*neg_a=*/false, A, /*neg_b=*/false, Bm,
            /*c_mod=*/(short)0, acc, /*reuse_a=*/false, /*reuse_b=*/false);
    }

    float* dst = inter + b * 256;
#pragma unroll
    for (int r = 0; r < 8; ++r) {
        const int m = r + 8*h;
        atomicAdd(&dst[m * 16 + mrow], acc[r]);  // integer-valued -> exact & deterministic
    }

    // Tail (N % 32) — scalar path, one wave per batch.
    if (blockIdx.x == 0 && waveInBlock == 0) {
        const int tail = N & 31;
        if (lane < tail) {
            const int n = (nChunks << 5) + lane;
            const int a1 = argmax16((const float4*)(p1 + (size_t)n * 16));
            const int a2 = argmax16((const float4*)(p2 + (size_t)n * 16));
            atomicAdd(&dst[a1 * 16 + a2], 1.0f);
        }
    }
}

// ---------------------------------------------------------------------------
// Kernel 2: Hungarian (minimize -iou) in f64; one active thread per (b, dir).
// dir==0 -> col2 (match mask2 cols to mask1 rows), dir==1 -> col1 (transposed iou).
// ---------------------------------------------------------------------------
__global__ void k_hungarian(const float* __restrict__ inter, int* __restrict__ cols) {
    if (threadIdx.x != 0) return;
    const int b   = blockIdx.x >> 1;
    const int dir = blockIdx.x & 1;
    const float* I = inter + b * 256;

    float rs[16], cs[16];
    for (int g = 0; g < 16; ++g) { rs[g] = 0.f; cs[g] = 0.f; }
    for (int g = 0; g < 16; ++g)
        for (int p = 0; p < 16; ++p) { float t = I[g*16 + p]; rs[g] += t; cs[p] += t; }

    double cost[16][16];
    for (int g = 0; g < 16; ++g)
        for (int p = 0; p < 16; ++p) {
            const float it  = I[g*16 + p];
            const float un  = fmaxf(rs[g] + cs[p] - it, 1e-10f);
            const float iou = it / un;                 // f32 division, like the reference
            if (dir == 0) cost[g][p] = -(double)iou;
            else          cost[p][g] = -(double)iou;   // transposed problem
        }

    const int n = 16;
    const double INF = 1e18;
    double u[17], v[17];
    int p[17], way[17];
    for (int j = 0; j <= n; ++j) { u[j] = 0.0; v[j] = 0.0; p[j] = 0; way[j] = 0; }

    for (int i = 1; i <= n; ++i) {
        p[0] = i;
        int j0 = 0;
        double minv[17];
        bool used[17];
        for (int j = 0; j <= n; ++j) { minv[j] = INF; used[j] = false; }
        do {
            used[j0] = true;
            const int i0 = p[j0];
            double delta = INF;
            int j1 = 0;
            for (int j = 1; j <= n; ++j) if (!used[j]) {
                const double cur = cost[i0 - 1][j - 1] - u[i0] - v[j];
                if (cur < minv[j]) { minv[j] = cur; way[j] = j0; }
                if (minv[j] < delta) { delta = minv[j]; j1 = j; }
            }
            for (int j = 0; j <= n; ++j) {
                if (used[j]) { u[p[j]] += delta; v[j] -= delta; }
                else          minv[j] -= delta;
            }
            j0 = j1;
        } while (p[j0] != 0);
        while (j0) { const int j1 = way[j0]; p[j0] = p[j1]; j0 = j1; }
    }

    int* out = cols + dir * (B_BATCH * 16) + b * 16;
    for (int j = 1; j <= n; ++j) out[p[j] - 1] = j - 1;
}

// ---------------------------------------------------------------------------
// Kernel 3: per-(b,n) loss terms + in-block reduction.
// ---------------------------------------------------------------------------
__device__ __forceinline__ float sel16(float4 r0, float4 r1, float4 r2, float4 r3, int c) {
    float4 q = (c < 8) ? ((c < 4) ? r0 : r1) : ((c < 12) ? r2 : r3);
    const int s = c & 3;
    return (s < 2) ? ((s == 0) ? q.x : q.y) : ((s == 2) ? q.z : q.w);
}

__global__ void k_loss(const float* __restrict__ m1, const float* __restrict__ m2,
                       const int* __restrict__ cols, float* __restrict__ partials, int N) {
    const int b = blockIdx.y;
    const int n = blockIdx.x * blockDim.x + threadIdx.x;

    __shared__ int   sc2[16], sc1[16];
    __shared__ float red[256];

    if (threadIdx.x < 16)      sc2[threadIdx.x]      = cols[b * 16 + threadIdx.x];
    else if (threadIdx.x < 32) sc1[threadIdx.x - 16] = cols[B_BATCH * 16 + b * 16 + threadIdx.x - 16];
    __syncthreads();

    float val = 0.f;
    if (n < N) {
        const size_t off = ((size_t)b * (size_t)N + (size_t)n) * 16;
        const float4* q1 = (const float4*)(m1 + off);
        const float4* q2 = (const float4*)(m2 + off);
        const float4 a0 = q1[0], a1 = q1[1], a2 = q1[2], a3 = q1[3];
        const float4 c0 = q2[0], c1 = q2[1], c2 = q2[2], c3 = q2[3];

        float d1 = 0.f;
#pragma unroll
        for (int i = 0; i < 16; ++i) {
            const float x = sel16(a0, a1, a2, a3, i);        // static -> folds to component
            const float y = sel16(c0, c1, c2, c3, sc2[i]);   // gather via cndmask tree
            const float t = x - y;
            d1 += t * t;
        }
        float d2 = 0.f;
#pragma unroll
        for (int i = 0; i < 16; ++i) {
            const float x = sel16(c0, c1, c2, c3, i);
            const float y = sel16(a0, a1, a2, a3, sc1[i]);
            const float t = x - y;
            d2 += t * t;
        }
        val = sqrtf(d1) + sqrtf(d2);
    }

    red[threadIdx.x] = val;
    __syncthreads();
#pragma unroll
    for (int s = 128; s > 0; s >>= 1) {
        if (threadIdx.x < s) red[threadIdx.x] += red[threadIdx.x + s];
        __syncthreads();
    }
    if (threadIdx.x == 0) partials[b * gridDim.x + blockIdx.x] = red[0];
}

// ---------------------------------------------------------------------------
// Kernel 4: deterministic f64 reduction of partials -> mean -> d_out[0].
// ---------------------------------------------------------------------------
__global__ void k_final(const float* __restrict__ partials, int nPart,
                        float* __restrict__ out, double invBN) {
    __shared__ double red[256];
    double s = 0.0;
    for (int i = threadIdx.x; i < nPart; i += blockDim.x) s += (double)partials[i];
    red[threadIdx.x] = s;
    __syncthreads();
#pragma unroll
    for (int st = 128; st > 0; st >>= 1) {
        if (threadIdx.x < st) red[threadIdx.x] += red[threadIdx.x + st];
        __syncthreads();
    }
    if (threadIdx.x == 0) out[0] = (float)(red[0] * invBN);
}

// ---------------------------------------------------------------------------
extern "C" void kernel_launch(void* const* d_in, const int* in_sizes, int n_in,
                              void* d_out, int out_size, void* d_ws, size_t ws_size,
                              hipStream_t stream) {
    const float* m1 = (const float*)d_in[0];
    const float* m2 = (const float*)d_in[1];
    const int B = B_BATCH;
    const int N = in_sizes[0] / (B * K_OBJ);

    // Workspace layout
    float* inter    = (float*)d_ws;                                        // B*256 f32
    int*   cols     = (int*)((char*)d_ws + (size_t)B * 256 * sizeof(float)); // 2*B*16 i32
    float* partials = (float*)((char*)cols + (size_t)2 * B * 16 * sizeof(int));

    // inter is accumulated atomically -> must start at zero every call
    hipMemsetAsync(inter, 0, (size_t)B * 256 * sizeof(float), stream);

    // 1) co-occurrence histogram via WMMA one-hot GEMM
    dim3 g1(32, B);
    k_inter<<<g1, 256, 0, stream>>>(m1, m2, inter, N);

    // 2) Hungarian (both directions), one active thread per problem
    k_hungarian<<<2 * B, 32, 0, stream>>>(inter, cols);

    // 3) loss terms + block reduction
    const int nb = (N + 255) / 256;
    dim3 g3(nb, B);
    k_loss<<<g3, 256, 0, stream>>>(m1, m2, cols, partials, N);

    // 4) final deterministic reduction
    k_final<<<1, 256, 0, stream>>>(partials, nb * B, (float*)d_out,
                                   1.0 / ((double)B * (double)N));
}
